// CausalMultiHeadedSelfAttention_85624468013196
// MI455X (gfx1250) — matrix operand order repaired and checked by reading, not compiled, not run on hardware
//
#include <hip/hip_runtime.h>
#include <hip/hip_bf16.h>

#define D_MODEL   1024
#define NUM_HEADS 16
#define HEAD_DIM  64
#define SEQ       2048
#define BATCH     4

typedef __attribute__((ext_vector_type(16))) _Float16 v16h;
typedef __attribute__((ext_vector_type(8)))  _Float16 v8h;
typedef __attribute__((ext_vector_type(8)))  float    v8f;

typedef union { v16h v; v8h h[2]; } frag16;

static __device__ __forceinline__ v8f wmma16x16x32(v16h a, v16h b, v8f c) {
    // D = A(16x32 f16) * B(32x16 f16) + C(16x16 f32)
    return __builtin_amdgcn_wmma_f32_16x16x32_f16(false, a, false, b,
                                                  (short)0, c, false, false);
}

// Async copy of 64 contiguous bytes global -> LDS (4 x b128, ASYNCcnt path).
// VDST = LDS byte address (low 32 bits of flat pointer), VADDR = global addr.
static __device__ __forceinline__ void async_copy64B(const _Float16* g,
                                                     _Float16* l) {
    const unsigned long long ga = (unsigned long long)(uintptr_t)g;
    const unsigned int       la = (unsigned int)(uintptr_t)l;
    asm volatile(
        "global_load_async_to_lds_b128 %0, %1, off\n\t"
        "global_load_async_to_lds_b128 %0, %1, off offset:16\n\t"
        "global_load_async_to_lds_b128 %0, %1, off offset:32\n\t"
        "global_load_async_to_lds_b128 %0, %1, off offset:48"
        :: "v"(la), "v"(ga) : "memory");
}
#define WAIT_ASYNCCNT(n) asm volatile("s_wait_asynccnt " #n ::: "memory")

// ---------------------------------------------------------------------------
// Kernel 0: one-time f32 -> f16 conversion (x and the four weight matrices).
// ---------------------------------------------------------------------------
__global__ void __launch_bounds__(256)
cvt_f16_kernel(const float* __restrict__ src, _Float16* __restrict__ dst, int n8)
{
    const int i = blockIdx.x * blockDim.x + threadIdx.x;
    if (i >= n8) return;
    const float4* s = (const float4*)src + (size_t)i * 2;
    float4 a = s[0], b = s[1];
    v8h o;
    o[0] = (_Float16)a.x; o[1] = (_Float16)a.y;
    o[2] = (_Float16)a.z; o[3] = (_Float16)a.w;
    o[4] = (_Float16)b.x; o[5] = (_Float16)b.y;
    o[6] = (_Float16)b.z; o[7] = (_Float16)b.w;
    *(v8h*)(dst + (size_t)i * 8) = o;
}

// ---------------------------------------------------------------------------
// Kernel 1: QKV projection (y = x * W^T) + fused RoPE (Q,K).
// Weight tile staged via double-buffered GLOBAL_LOAD_ASYNC_TO_LDS_B128.
// Q,K stored [B,H,S,64]; V stored TRANSPOSED [B,H,64,S].
// Block = 128 threads (4 waves); block tile 64(M) x 128(N); wave = 16x128.
// ---------------------------------------------------------------------------
__global__ void __launch_bounds__(128)
qkv_rope_kernel(const _Float16* __restrict__ Xh,
                const _Float16* __restrict__ Wh,   // [3][1024][1024] f16
                const float* __restrict__ cosT,
                const float* __restrict__ sinT,
                _Float16* __restrict__ Qh,
                _Float16* __restrict__ Kh,
                _Float16* __restrict__ Vt)
{
    __shared__ _Float16 bstage[2][128][32];   // 16 KB double-buffered tile

    const int widx = blockIdx.z;
    const _Float16* w   = Wh + (size_t)widx * D_MODEL * D_MODEL;
    _Float16*       dst = (widx == 0) ? Qh : (widx == 1) ? Kh : Vt;

    const int tid  = threadIdx.x;
    const int lane = tid & 31;
    const int wave = tid >> 5;
    const int half = lane >> 4;
    const int l15  = lane & 15;
    const int Mw   = blockIdx.y * 64 + wave * 16;
    const int Nb   = blockIdx.x * 128;

    const _Float16* xrow  = Xh + (size_t)(Mw + l15) * D_MODEL;
    const _Float16* wload = w + (size_t)(Nb + tid) * D_MODEL;  // row staged by this thread

    v8f acc[8] = {};

    async_copy64B(wload, &bstage[0][tid][0]);          // preload tile 0

    for (int i = 0; i < D_MODEL / 32; ++i) {
        const int k0 = i * 32;
        if (i + 1 < D_MODEL / 32)                       // issue next tile
            async_copy64B(wload + k0 + 32, &bstage[(i + 1) & 1][tid][0]);

        frag16 af;                                      // A fragment from global
        af.h[0] = *(const v8h*)(xrow + k0 + half * 8);
        af.h[1] = *(const v8h*)(xrow + k0 + half * 8 + 16);

        if (i + 1 < D_MODEL / 32) { WAIT_ASYNCCNT(4); } // current tile landed,
        else                      { WAIT_ASYNCCNT(0); } // next still in flight
        __syncthreads();                                // all threads' data visible

        const _Float16 (*bs)[32] = bstage[i & 1];
        #pragma unroll
        for (int t = 0; t < 8; ++t) {
            const int nl = t * 16 + l15;
            frag16 bf;
            bf.h[0] = *(const v8h*)&bs[nl][half * 8];
            bf.h[1] = *(const v8h*)&bs[nl][half * 8 + 16];
            acc[t] = wmma16x16x32(af.v, bf.v, acc[t]);
        }
        __syncthreads();                                // tile consumed -> reusable
    }

    // ---- Epilogue: RoPE (q,k) in D-layout; V stored transposed ----
    #pragma unroll
    for (int t = 0; t < 8; ++t) {
        const int n  = Nb + t * 16 + l15;
        const int hh = n >> 6;
        const int dh = n & 63;
        const int fi = (n & 63) >> 1;
        #pragma unroll
        for (int r = 0; r < 8; ++r) {
            const int mrow = Mw + r + half * 8;
            const int bb   = mrow >> 11;
            const int s    = mrow & (SEQ - 1);
            float val = acc[t][r];
            if (widx < 2) {
                float partner = __shfl_xor(val, 1, 32);
                float c  = cosT[s * 32 + fi];
                float sn = sinT[s * 32 + fi];
                float o = (n & 1) ? (val * c + partner * sn)
                                  : (val * c - partner * sn);
                dst[((size_t)((bb * NUM_HEADS + hh) * SEQ + s)) * HEAD_DIM + dh] =
                    (_Float16)o;
            } else {
                // V transposed: [B, H, dh, s]
                dst[((size_t)((bb * NUM_HEADS + hh) * HEAD_DIM + dh)) * SEQ + s] =
                    (_Float16)val;
            }
        }
    }
}

// ---------------------------------------------------------------------------
// Kernel 2: causal flash attention, f16 in/out, f32 accum + online softmax.
// K/V tiles cooperatively staged in double-buffered LDS via async-to-LDS.
// Grid: (S/64, H, B), block 128 (4 waves; each wave owns 16 query rows).
// Loop over key blocks is BLOCK-uniform (required for barriers); waves whose
// rows are fully masked in trailing blocks are no-ops in the online softmax.
// ---------------------------------------------------------------------------
__global__ void __launch_bounds__(128)
attn_kernel(const _Float16* __restrict__ Qh,
            const _Float16* __restrict__ Kh,
            const _Float16* __restrict__ Vt,
            _Float16* __restrict__ Oh)
{
    __shared__ _Float16 kstage[2][32][64];   // 8 KB: 32 keys x 64 dims
    __shared__ _Float16 vstage[2][64][32];   // 8 KB: 64 dims x 32 keys (V^T)
    __shared__ _Float16 plds[4][16][32];     // 4 KB: per-wave P tile

    const int tid  = threadIdx.x;
    const int lane = tid & 31;
    const int wave = tid >> 5;
    const int half = lane >> 4;
    const int l15  = lane & 15;
    const int h    = blockIdx.y;
    const int b    = blockIdx.z;
    const int q0   = blockIdx.x * 64 + wave * 16;

    const size_t headoff  = (size_t)(b * NUM_HEADS + h) * SEQ * HEAD_DIM;
    const _Float16* kbaseptr = Kh + headoff;                 // [key][64] rows
    const _Float16* vbaseptr = Vt + headoff;                 // [dim][SEQ] rows
    const int    koff     = half * 8;

    frag16 aq0, aq1;
    {
        const _Float16* qrow = Qh + headoff + (size_t)(q0 + l15) * HEAD_DIM;
        aq0.h[0] = *(const v8h*)(qrow + koff);
        aq0.h[1] = *(const v8h*)(qrow + koff + 16);
        aq1.h[0] = *(const v8h*)(qrow + 32 + koff);
        aq1.h[1] = *(const v8h*)(qrow + 32 + koff + 16);
    }

    float mrow[8], lrow[8];
    #pragma unroll
    for (int r = 0; r < 8; ++r) { mrow[r] = -__builtin_inff(); lrow[r] = 0.f; }
    v8f ao[4] = {};

    // block-uniform causal bound: keys <= blockIdx.x*64 + 63
    const int nkb = (blockIdx.x * 64 + 95) / 32;

    // stage tile 0: threads 0-63 copy K block (contiguous 4 KB),
    //               threads 64-127 copy one V^T row (64 B) each.
    if (tid < 64)
        async_copy64B(kbaseptr + tid * 32, &kstage[0][0][0] + tid * 32);
    else
        async_copy64B(vbaseptr + (size_t)(tid - 64) * SEQ,
                      &vstage[0][tid - 64][0]);

    for (int kb = 0; kb < nkb; ++kb) {
        const int kbase = kb * 32;

        if (kb + 1 < nkb) {                       // issue next K/V tile
            const int knext = kbase + 32;
            if (tid < 64)
                async_copy64B(kbaseptr + (size_t)knext * HEAD_DIM + tid * 32,
                              &kstage[(kb + 1) & 1][0][0] + tid * 32);
            else
                async_copy64B(vbaseptr + (size_t)(tid - 64) * SEQ + knext,
                              &vstage[(kb + 1) & 1][tid - 64][0]);
            WAIT_ASYNCCNT(4);                     // current tile landed
        } else {
            WAIT_ASYNCCNT(0);
        }
        __syncthreads();

        const _Float16 (*ks)[64] = kstage[kb & 1];
        const _Float16 (*vs)[32] = vstage[kb & 1];

        // ---- scores: S = Q * K^T (two 16-key tiles, head-dim in 2 steps) ----
        v8f s0 = {}, s1 = {};
        #pragma unroll
        for (int t = 0; t < 2; ++t) {
            const _Float16* krow = &ks[t * 16 + l15][0];
            frag16 b0, b1;
            b0.h[0] = *(const v8h*)(krow + half * 8);
            b0.h[1] = *(const v8h*)(krow + half * 8 + 16);
            b1.h[0] = *(const v8h*)(krow + 32 + half * 8);
            b1.h[1] = *(const v8h*)(krow + 32 + half * 8 + 16);
            v8f s = {};
            s = wmma16x16x32(aq0.v, b0.v, s);
            s = wmma16x16x32(aq1.v, b1.v, s);
            if (t == 0) s0 = s; else s1 = s;
        }

        // ---- causal mask + online softmax (rows live across 16 lanes) ----
        #pragma unroll
        for (int r = 0; r < 8; ++r) {
            const int qi = q0 + r + half * 8;
            float x0 = s0[r] * 0.125f;
            float x1 = s1[r] * 0.125f;
            if (kbase + l15 > qi)      x0 = -__builtin_inff();
            if (kbase + 16 + l15 > qi) x1 = -__builtin_inff();

            float mx = fmaxf(x0, x1);
            #pragma unroll
            for (int off = 1; off < 16; off <<= 1)
                mx = fmaxf(mx, __shfl_xor(mx, off, 32));

            const float mnew  = fmaxf(mrow[r], mx);
            const float alpha = __expf(mrow[r] - mnew);
            const float p0    = __expf(x0 - mnew);
            const float p1    = __expf(x1 - mnew);

            float rs = p0 + p1;
            #pragma unroll
            for (int off = 1; off < 16; off <<= 1)
                rs += __shfl_xor(rs, off, 32);

            lrow[r] = lrow[r] * alpha + rs;
            mrow[r] = mnew;
            #pragma unroll
            for (int t = 0; t < 4; ++t) ao[t][r] *= alpha;

            plds[wave][r + half * 8][l15]      = (_Float16)p0;
            plds[wave][r + half * 8][16 + l15] = (_Float16)p1;
        }

        // ---- P via LDS round-trip (D-layout -> A-layout transpose) ----
        frag16 ap;
        {
            const _Float16* prow = &plds[wave][l15][0];
            ap.h[0] = *(const v8h*)(prow + koff);
            ap.h[1] = *(const v8h*)(prow + koff + 16);
        }

        // ---- O += P * V : V^T rows contiguous in key dim (in LDS) ----
        #pragma unroll
        for (int t = 0; t < 4; ++t) {
            const _Float16* vrow = &vs[t * 16 + l15][0];
            frag16 bv;
            bv.h[0] = *(const v8h*)(vrow + half * 8);
            bv.h[1] = *(const v8h*)(vrow + half * 8 + 16);
            ao[t] = wmma16x16x32(ap.v, bv.v, ao[t]);
        }
        __syncthreads();                          // tile consumed -> reusable
    }

    // ---- normalize and store f16 [B, S, D] ----
    #pragma unroll
    for (int t = 0; t < 4; ++t) {
        #pragma unroll
        for (int r = 0; r < 8; ++r) {
            const int s = q0 + r + half * 8;
            float o = ao[t][r] / lrow[r];
            Oh[(size_t)(b * SEQ + s) * D_MODEL + h * 64 + t * 16 + l15] =
                (_Float16)o;
        }
    }
}

// ---------------------------------------------------------------------------
// Kernel 3: output projection out = O * Wo^T, f16 x f16 -> f32.
// Same blocking as kernel 1 (async double-buffered weight staging).
// ---------------------------------------------------------------------------
__global__ void __launch_bounds__(128)
out_proj_kernel(const _Float16* __restrict__ Oh,
                const _Float16* __restrict__ Who,
                float* __restrict__ out)
{
    __shared__ _Float16 bstage[2][128][32];

    const int tid  = threadIdx.x;
    const int lane = tid & 31;
    const int wave = tid >> 5;
    const int half = lane >> 4;
    const int l15  = lane & 15;
    const int Mw   = blockIdx.y * 64 + wave * 16;
    const int Nb   = blockIdx.x * 128;

    const _Float16* orow  = Oh + (size_t)(Mw + l15) * D_MODEL;
    const _Float16* wload = Who + (size_t)(Nb + tid) * D_MODEL;

    v8f acc[8] = {};

    async_copy64B(wload, &bstage[0][tid][0]);

    for (int i = 0; i < D_MODEL / 32; ++i) {
        const int k0 = i * 32;
        if (i + 1 < D_MODEL / 32)
            async_copy64B(wload + k0 + 32, &bstage[(i + 1) & 1][tid][0]);

        frag16 af;
        af.h[0] = *(const v8h*)(orow + k0 + half * 8);
        af.h[1] = *(const v8h*)(orow + k0 + half * 8 + 16);

        if (i + 1 < D_MODEL / 32) { WAIT_ASYNCCNT(4); }
        else                      { WAIT_ASYNCCNT(0); }
        __syncthreads();

        const _Float16 (*bs)[32] = bstage[i & 1];
        #pragma unroll
        for (int t = 0; t < 8; ++t) {
            const int nl = t * 16 + l15;
            frag16 bf;
            bf.h[0] = *(const v8h*)&bs[nl][half * 8];
            bf.h[1] = *(const v8h*)&bs[nl][half * 8 + 16];
            acc[t] = wmma16x16x32(af.v, bf.v, acc[t]);
        }
        __syncthreads();
    }

    #pragma unroll
    for (int t = 0; t < 8; ++t) {
        #pragma unroll
        for (int r = 0; r < 8; ++r) {
            out[(size_t)(Mw + r + half * 8) * D_MODEL + Nb + t * 16 + l15] =
                acc[t][r];
        }
    }
}

// ---------------------------------------------------------------------------
extern "C" void kernel_launch(void* const* d_in, const int* in_sizes, int n_in,
                              void* d_out, int out_size, void* d_ws, size_t ws_size,
                              hipStream_t stream)
{
    (void)in_sizes; (void)n_in; (void)out_size; (void)ws_size;

    const float* x    = (const float*)d_in[0];
    const float* wq   = (const float*)d_in[1];
    const float* wk   = (const float*)d_in[2];
    const float* wv   = (const float*)d_in[3];
    const float* wo   = (const float*)d_in[4];
    const float* cosT = (const float*)d_in[5];
    const float* sinT = (const float*)d_in[6];
    float*       out  = (float*)d_out;

    const size_t xElems = (size_t)BATCH * SEQ * D_MODEL;        // 8.4M
    const size_t wElems = (size_t)D_MODEL * D_MODEL;            // 1.05M

    _Float16* Xh  = (_Float16*)d_ws;
    _Float16* Wh  = Xh + xElems;          // [wq | wk | wv] then wo
    _Float16* Who = Wh + 3 * wElems;
    _Float16* Qh  = Who + wElems;
    _Float16* Kh  = Qh + xElems;
    _Float16* Vt  = Kh + xElems;          // [B, H, 64, S]
    _Float16* Oh  = Vt + xElems;          // [B, S, D]

    // ---- stage 0: convert everything to f16 once ----
    {
        int n8 = (int)(xElems / 8);
        cvt_f16_kernel<<<(n8 + 255) / 256, 256, 0, stream>>>(x, Xh, n8);
        n8 = (int)(wElems / 8);
        cvt_f16_kernel<<<(n8 + 255) / 256, 256, 0, stream>>>(wq, Wh + 0 * wElems, n8);
        cvt_f16_kernel<<<(n8 + 255) / 256, 256, 0, stream>>>(wk, Wh + 1 * wElems, n8);
        cvt_f16_kernel<<<(n8 + 255) / 256, 256, 0, stream>>>(wv, Wh + 2 * wElems, n8);
        cvt_f16_kernel<<<(n8 + 255) / 256, 256, 0, stream>>>(wo, Who, n8);
    }

    dim3 blk(128);
    qkv_rope_kernel<<<dim3(D_MODEL / 128, (BATCH * SEQ) / 64, 3), blk, 0, stream>>>(
        Xh, Wh, cosT, sinT, Qh, Kh, Vt);
    attn_kernel<<<dim3(SEQ / 64, NUM_HEADS, BATCH), blk, 0, stream>>>(
        Qh, Kh, Vt, Oh);
    out_proj_kernel<<<dim3(D_MODEL / 128, (BATCH * SEQ) / 64, 1), blk, 0, stream>>>(
        Oh, Who, out);
}
